// Model_15504831939020
// MI455X (gfx1250) — compile-verified, matrix operand order checked by reading, not run on hardware
//
#include <hip/hip_runtime.h>

typedef _Float16 h4   __attribute__((ext_vector_type(4)));
typedef _Float16 v8h  __attribute__((ext_vector_type(8)));
typedef _Float16 v16h __attribute__((ext_vector_type(16)));
typedef float    v8f  __attribute__((ext_vector_type(8)));
typedef int      v4i  __attribute__((ext_vector_type(4)));

#define TSTEPS 100
#define B_ 8
#define S_ 512
#define F_ 128
#define H_ 512
#define M_ (B_ * S_)   // 4096 rows

// gfx1250 async global->LDS DMA path (ASYNCcnt), with safe fallback.
#if __has_builtin(__builtin_amdgcn_global_load_async_to_lds_b128)
#define HAVE_ASYNC_LDS 1
#else
#define HAVE_ASYNC_LDS 0
#endif

__device__ __forceinline__ float softplusf(float v) {
    // numerically stable log1p(exp(v))
    return fmaxf(v, 0.0f) + log1pf(__expf(-fabsf(v)));
}

__device__ __forceinline__ v16h cat8(v8h lo, v8h hi) {
    v16h r;
#pragma unroll
    for (int i = 0; i < 8; ++i) { r[i] = lo[i]; r[8 + i] = hi[i]; }
    return r;
}

// A fragment: lane holds row m=lane&15, K-half kb=(lane>>4)*8:
// halfs [kb..kb+7] and [kb+16..kb+23] of the 32-wide k-step (p pre-offset by kb).
__device__ __forceinline__ v16h load_af(const _Float16* p) {
    return cat8(*(const v8h*)p, *(const v8h*)(p + 16));
}

// B fragment from LDS slab: tile (kt,j) is 512 contiguous halfs, lane-major.
__device__ __forceinline__ v16h load_bf(const _Float16* lds, int kt, int j, int lane) {
    const _Float16* p = lds + ((kt * 4 + j) << 9) + lane * 16;
    return cat8(*(const v8h*)p, *(const v8h*)(p + 8));
}

__device__ __forceinline__ v8f wmma_f16(v16h a, v16h b, v8f c) {
    return __builtin_amdgcn_wmma_f32_16x16x32_f16(false, a, false, b,
                                                  (short)0, c, false, false);
}

// ---------------------------------------------------------------------------
// xt[b,s,f] = x[b,s,f] + sum_{j<t[b]} noise[j,b,s,f]  -> stored as f16 [M_,F_]
// ---------------------------------------------------------------------------
__global__ void noisy_prefix_kernel(const float* __restrict__ x,
                                    const float* __restrict__ noise,
                                    const int* __restrict__ t,
                                    _Float16* __restrict__ xt16)
{
    const int SF4 = S_ * F_ / 4;                       // float4 per batch
    int idx = blockIdx.x * blockDim.x + threadIdx.x;   // [0, B*S*F/4)
    int b = idx / SF4;
    int tb = t[b];

    float4 acc = *(const float4*)(x + (size_t)idx * 4);
    const float* np = noise + (size_t)idx * 4;
    const size_t stride = (size_t)B_ * S_ * F_;

    for (int j = 0; j < tb; ++j) {
        const float* p = np + (size_t)j * stride;
        if (j + 2 < tb) __builtin_prefetch(p + 2 * stride, 0, 0);
        float4 nv = *(const float4*)p;
        acc.x += nv.x; acc.y += nv.y; acc.z += nv.z; acc.w += nv.w;
    }
    h4 o;
    o[0] = (_Float16)acc.x; o[1] = (_Float16)acc.y;
    o[2] = (_Float16)acc.z; o[3] = (_Float16)acc.w;
    *(h4*)(xt16 + (size_t)idx * 4) = o;
}

// ---------------------------------------------------------------------------
// Pack f32 weight [K,N] (row-major) into f16 WMMA B-fragment tiles:
// tile (kt,nt) is 32x16; element e = lane*16 + j, lane holds column n = nt*16
// + (lane&15); K-half kb = (lane>>4)*8; j<8 -> k=kb+j, j>=8 -> k=kb+16+(j-8).
// ---------------------------------------------------------------------------
__global__ void pack_weight_kernel(const float* __restrict__ W,
                                   _Float16* __restrict__ Wp,
                                   int K, int N)
{
    int idx = blockIdx.x * blockDim.x + threadIdx.x;
    if (idx >= K * N) return;
    int e    = idx & 511;
    int tile = idx >> 9;
    int ntiles = N >> 4;
    int kt = tile / ntiles;
    int nt = tile - kt * ntiles;
    int lane = e >> 4;
    int j    = e & 15;
    int kb   = (lane >> 4) << 3;
    int kl   = (j < 8) ? (kb + j) : (kb + 16 + (j - 8));
    int n = nt * 16 + (lane & 15);
    int k = kt * 32 + kl;
    Wp[idx] = (_Float16)W[(size_t)k * N + n];
}

// ---------------------------------------------------------------------------
// C[row,col] = act( (A @ W) * gamma[t[row/S]][col] + bias[col] )
// A: [M_,K] f16 row-major.  Wp: packed fragment tiles [K/32][N/16][512].
// Block = 8 waves, block tile 128x64. B slab (shared by all 8 waves) is
// DMA'd into LDS once (async-to-LDS), then the k-loop is register
// double-buffered: next A (global) + next 4 B frags (ds_load) are issued
// before the 4 WMMAs consuming current fragments -> no per-WMMA full waits.
// mode 0: softplus -> f16 out16     mode 1: linear -> f32 out32
// ---------------------------------------------------------------------------
__global__ void __launch_bounds__(256)
wmma_gemm_kernel(const _Float16* __restrict__ A,
                 const _Float16* __restrict__ Wp,
                 const float* __restrict__ bias,
                 const float* __restrict__ gamma,  // or nullptr
                 const int* __restrict__ t,
                 _Float16* __restrict__ out16,
                 float* __restrict__ out32,
                 int K, int N, int mode)
{
    __shared__ _Float16 Blds[16 * 2048];   // 64 KB max (K=512: 16 k-steps * 4KB)

    const int tid  = threadIdx.x;
    const int lane = tid & 31;
    const int wave = tid >> 5;
    const int ksteps = K >> 5;
    const int ntiles = N >> 4;
    const int nt0 = blockIdx.x << 2;

    // ---- cooperative B-slab prefetch: ksteps rows of 4KB (256 lanes x 16B) ----
    for (int kt = 0; kt < ksteps; ++kt) {
        const _Float16* src = Wp + ((size_t)(kt * ntiles + nt0) << 9) + tid * 8;
        _Float16* dst = &Blds[(kt << 11) + tid * 8];
#if HAVE_ASYNC_LDS
        __builtin_amdgcn_global_load_async_to_lds_b128(
            (__attribute__((address_space(1))) v4i*)src,
            (__attribute__((address_space(3))) v4i*)dst, 0, 0);
#else
        *(v8h*)dst = *(const v8h*)src;
#endif
    }
#if HAVE_ASYNC_LDS
#if __has_builtin(__builtin_amdgcn_s_wait_asynccnt)
    __builtin_amdgcn_s_wait_asynccnt(0);
#else
    asm volatile("s_wait_asynccnt 0" ::: "memory");
#endif
#endif
    __syncthreads();

    // ---- main GEMM loop, register double-buffered over k-steps ----
    const int m0   = ((int)blockIdx.y * 8 + wave) * 16;
    const int mrow = lane & 15;
    const int kbo  = (lane >> 4) << 3;
    const _Float16* arow = A + (size_t)(m0 + mrow) * K + kbo;

    v8f acc[4] = {};

    v16h afC = load_af(arow);
    v16h bC0 = load_bf(Blds, 0, 0, lane);
    v16h bC1 = load_bf(Blds, 0, 1, lane);
    v16h bC2 = load_bf(Blds, 0, 2, lane);
    v16h bC3 = load_bf(Blds, 0, 3, lane);

    for (int kt = 0; kt + 1 < ksteps; ++kt) {
        v16h afN = load_af(arow + (kt + 1) * 32);
        v16h bN0 = load_bf(Blds, kt + 1, 0, lane);
        v16h bN1 = load_bf(Blds, kt + 1, 1, lane);
        v16h bN2 = load_bf(Blds, kt + 1, 2, lane);
        v16h bN3 = load_bf(Blds, kt + 1, 3, lane);

        acc[0] = wmma_f16(afC, bC0, acc[0]);
        acc[1] = wmma_f16(afC, bC1, acc[1]);
        acc[2] = wmma_f16(afC, bC2, acc[2]);
        acc[3] = wmma_f16(afC, bC3, acc[3]);

        afC = afN;
        bC0 = bN0; bC1 = bN1; bC2 = bN2; bC3 = bN3;
    }
    acc[0] = wmma_f16(afC, bC0, acc[0]);
    acc[1] = wmma_f16(afC, bC1, acc[1]);
    acc[2] = wmma_f16(afC, bC2, acc[2]);
    acc[3] = wmma_f16(afC, bC3, acc[3]);

    // ---- epilogue: gamma * acc + bias, activation, store ----
    const int tb   = (gamma != nullptr) ? t[m0 / S_] : 0;
    const int rofs = (lane >> 4) << 3;
#pragma unroll
    for (int j = 0; j < 4; ++j) {
        int col = (nt0 + j) * 16 + (lane & 15);
        float gm = (gamma != nullptr) ? gamma[(size_t)tb * N + col] : 1.0f;
        float bi = bias[col];
#pragma unroll
        for (int r = 0; r < 8; ++r) {
            int row = m0 + rofs + r;
            float v = acc[j][r] * gm + bi;
            if (mode == 0) {
                out16[(size_t)row * N + col] = (_Float16)softplusf(v);
            } else {
                out32[(size_t)row * N + col] = v;
            }
        }
    }
}

// ---------------------------------------------------------------------------
// pooled[b,h] = mean_s ht2[b,s,h];  pred_t[b] = sigmoid(pooled . wt3 + bt3)
// also writes gt_t[b] = t[b]/TSTEPS
// ---------------------------------------------------------------------------
__global__ void pool_head_kernel(const _Float16* __restrict__ ht2,
                                 const float* __restrict__ wt3,
                                 const float* __restrict__ bt3,
                                 const int* __restrict__ t,
                                 float* __restrict__ out)
{
    const int b = blockIdx.x;
    const int tid = threadIdx.x;   // 256 threads
    float partial = 0.0f;
    for (int h = tid; h < H_; h += 256) {
        float sum = 0.0f;
        for (int s = 0; s < S_; ++s)
            sum += (float)ht2[((size_t)(b * S_ + s)) * H_ + h];
        partial += (sum * (1.0f / (float)S_)) * wt3[h];
    }
    __shared__ float red[256];
    red[tid] = partial;
    __syncthreads();
    for (int off = 128; off > 0; off >>= 1) {
        if (tid < off) red[tid] += red[tid + off];
        __syncthreads();
    }
    if (tid == 0) {
        float z = red[0] + bt3[0];
        out[(size_t)M_ * F_ + b]     = 1.0f / (1.0f + __expf(-z));     // pred_t
        out[(size_t)M_ * F_ + 8 + b] = (float)t[b] / (float)TSTEPS;   // gt_t
    }
}

// ---------------------------------------------------------------------------
extern "C" void kernel_launch(void* const* d_in, const int* in_sizes, int n_in,
                              void* d_out, int out_size, void* d_ws, size_t ws_size,
                              hipStream_t stream)
{
    (void)in_sizes; (void)n_in; (void)out_size; (void)ws_size;

    const float* x      = (const float*)d_in[0];
    const float* noise  = (const float*)d_in[1];
    const int*   t      = (const int*)d_in[2];
    const float* gamma1 = (const float*)d_in[3];
    const float* W1     = (const float*)d_in[4];
    const float* b1     = (const float*)d_in[5];
    const float* gamma2 = (const float*)d_in[6];
    const float* W2     = (const float*)d_in[7];
    const float* b2     = (const float*)d_in[8];
    const float* W3     = (const float*)d_in[9];
    const float* b3     = (const float*)d_in[10];
    const float* Wt1    = (const float*)d_in[11];
    const float* bt1    = (const float*)d_in[12];
    const float* Wt2    = (const float*)d_in[13];
    const float* bt2    = (const float*)d_in[14];
    const float* wt3    = (const float*)d_in[15];
    const float* bt3    = (const float*)d_in[16];
    float* out = (float*)d_out;

    char* ws = (char*)d_ws;
    _Float16* xt16 = (_Float16*)(ws + 0x0000000);  // 4096*128*2 = 1 MB
    _Float16* h1   = (_Float16*)(ws + 0x0100000);  // 4096*512*2 = 4 MB
    _Float16* h2   = (_Float16*)(ws + 0x0500000);  // 4 MB
    _Float16* ht1  = (_Float16*)(ws + 0x0900000);  // 4 MB
    _Float16* ht2  = (_Float16*)(ws + 0x0D00000);  // 4 MB
    _Float16* W1p  = (_Float16*)(ws + 0x1100000);  // 128 KB
    _Float16* W2p  = (_Float16*)(ws + 0x1120000);  // 512 KB
    _Float16* W3p  = (_Float16*)(ws + 0x11A0000);  // 128 KB
    _Float16* Wt1p = (_Float16*)(ws + 0x11C0000);  // 128 KB
    _Float16* Wt2p = (_Float16*)(ws + 0x11E0000);  // 512 KB

    // 1) prefix-sum gather + f16 convert (HBM-bound phase)
    noisy_prefix_kernel<<<dim3((B_ * S_ * F_ / 4) / 256), 256, 0, stream>>>(
        x, noise, t, xt16);

    // 2) weight packing (tiny, once per call)
    pack_weight_kernel<<<dim3(F_ * H_ / 256), 256, 0, stream>>>(W1,  W1p,  F_, H_);
    pack_weight_kernel<<<dim3(H_ * H_ / 256), 256, 0, stream>>>(W2,  W2p,  H_, H_);
    pack_weight_kernel<<<dim3(H_ * F_ / 256), 256, 0, stream>>>(W3,  W3p,  H_, F_);
    pack_weight_kernel<<<dim3(F_ * H_ / 256), 256, 0, stream>>>(Wt1, Wt1p, F_, H_);
    pack_weight_kernel<<<dim3(H_ * H_ / 256), 256, 0, stream>>>(Wt2, Wt2p, H_, H_);

    // 3) WMMA GEMM pipeline
    dim3 g512(H_ / 64, M_ / 128);   // N=512 tiles
    dim3 g128(F_ / 64, M_ / 128);   // N=128 tiles
    // conditional model (gamma-modulated)
    wmma_gemm_kernel<<<g512, 256, 0, stream>>>(xt16, W1p, b1, gamma1, t,
                                               h1, nullptr, F_, H_, 0);
    wmma_gemm_kernel<<<g512, 256, 0, stream>>>(h1, W2p, b2, gamma2, t,
                                               h2, nullptr, H_, H_, 0);
    wmma_gemm_kernel<<<g128, 256, 0, stream>>>(h2, W3p, b3, nullptr, nullptr,
                                               nullptr, out, H_, F_, 1);
    // conditional model_t
    wmma_gemm_kernel<<<g512, 256, 0, stream>>>(xt16, Wt1p, bt1, nullptr, nullptr,
                                               ht1, nullptr, F_, H_, 0);
    wmma_gemm_kernel<<<g512, 256, 0, stream>>>(ht1, Wt2p, bt2, nullptr, nullptr,
                                               ht2, nullptr, H_, H_, 0);

    // 4) mean-pool + sigmoid head + gt_t
    pool_head_kernel<<<dim3(B_), 256, 0, stream>>>(ht2, wt3, bt3, t, out);
}
